// QuantLinear_1717986918631
// MI455X (gfx1250) — matrix operand order repaired and checked by reading, not compiled, not run on hardware
//
#include <hip/hip_runtime.h>
#include <hip/hip_bf16.h>

typedef __attribute__((ext_vector_type(16))) _Float16 v16h;
typedef __attribute__((ext_vector_type(8)))  float    v8f;
typedef __attribute__((ext_vector_type(2)))  _Float16 h2;

#define K_DIM   4096
#define N_DIM   11008
#define M_DIM   16
#define GSIZE   128
#define NGROUP  (K_DIM / GSIZE)          // 32
#define KW      (K_DIM / 8)              // 512 packed rows
#define NZ      (N_DIM / 8)              // 1376 packed zero cols
#define NTILES  (K_DIM / 32)             // 128 k-tiles of 32

// ---------------------------------------------------------------------------
// K-permutation shared by A and B fragments.
// WMMA position p (0..31) -> actual k offset:  p = 8b + 2i + j  =>  k = 8b + i + 4j
// This makes B dequant a single (shift+)AND+OR+PK_FMA per f16 pair, because
// ((word >> 4i) & 0x000F000F) naturally yields the (q_i, q_{i+4}) pair.
// ---------------------------------------------------------------------------

// Pre-pack x (f32, [16][4096]) into f16 WMMA-A fragments with the permutation
// baked in. Apack layout: [tile t][lane][16 halves] contiguous (32 B / lane).
__global__ __launch_bounds__(128) void apack_kernel(const float* __restrict__ x,
                                                    _Float16* __restrict__ Apack) {
    int idx = blockIdx.x * blockDim.x + threadIdx.x;   // 0 .. 128*32-1
    if (idx >= NTILES * 32) return;
    int t    = idx >> 5;
    int lane = idx & 31;
    int lh   = lane >> 4;      // lane half
    int m    = lane & 15;      // A-matrix row

    v16h vals;
#pragma unroll
    for (int v = 0; v < 8; ++v) {
#pragma unroll
        for (int h = 0; h < 2; ++h) {
            // A layout (matches B): lanes 0-15 VGPR0-7 -> pos 0..15;
            // lanes 16-31 add 16.
            int pos = 16 * lh + 2 * v + h;
            int b = pos >> 3, rp = pos & 7;
            int k = 8 * b + (rp >> 1) + 4 * (rp & 1);   // permutation pi(pos)
            vals[2 * v + h] = (_Float16)x[(size_t)m * K_DIM + (size_t)t * 32 + k];
        }
    }
    *(v16h*)(Apack + (size_t)idx * 16) = vals;
}

// Main kernel: one workgroup (8 waves) per 16-column tile; waves split K 8-ways.
__global__ __launch_bounds__(256) void qgemm_wmma(
    const _Float16* __restrict__ Apack,   // [128][32][16] f16
    const int*      __restrict__ qweight, // [512][11008]
    const float*    __restrict__ scales,  // [32][11008]
    const int*      __restrict__ qzeros,  // [32][1376]
    float*          __restrict__ out)     // [16][11008]
{
    __shared__ float red[8 * 256];        // 8 KB: per-wave partial 16x16 f32 tiles

    const int tid  = threadIdx.x;
    const int wave = tid >> 5;            // 0..7 : K-split index
    const int lane = tid & 31;
    const int lh   = lane >> 4;
    const int ln   = lane & 15;
    const int n0   = blockIdx.x * 16;
    const int n    = n0 + ln;

    v8f acc = {};
    const int g0 = wave * 4;              // 4 quant groups (128 k each) per wave

#pragma unroll 1
    for (int gi = 0; gi < 4; ++gi) {
        const int g = g0 + gi;
        const float s  = scales[(size_t)g * N_DIM + n];
        const int  zw  = qzeros[(size_t)g * NZ + (n >> 3)];
        const int  z   = (zw >> ((n & 7) * 4)) & 0xF;
        // dequant constants: w = s*(q - z) = s*((1024+q) - (1024+z))
        const _Float16 hs  = (_Float16)s;
        const _Float16 hbs = (_Float16)((float)(z + 1024) * s);
        const h2 hs2  = { hs,  hs  };
        const h2 nbs2 = { (_Float16)(-hbs), (_Float16)(-hbs) };

        const int t0 = g * 4;             // k-tile index of this group
#pragma unroll
        for (int ti = 0; ti < 4; ++ti) {
            const int t = t0 + ti;

            // A fragment: 32 contiguous bytes per lane (L2-resident)
            v16h a = *(const v16h*)(Apack + ((size_t)(t * 32 + lane)) * 16);

            // B: two packed-int32 rows per lane (read-once -> non-temporal)
            const int r = (t << 2) + (lh << 1);   // k0/8 + 2*lh
            const unsigned w0 =
                (unsigned)__builtin_nontemporal_load(qweight + (size_t)r * N_DIM + n);
            const unsigned w1 =
                (unsigned)__builtin_nontemporal_load(qweight + (size_t)(r + 1) * N_DIM + n);

            v16h b;
#pragma unroll
            for (int v = 0; v < 8; ++v) {
                unsigned src  = (v < 4) ? w0 : w1;
                unsigned bits = ((src >> ((v & 3) * 4)) & 0x000F000Fu) | 0x64006400u;
                h2 hq  = __builtin_bit_cast(h2, bits);
                h2 res = hq * hs2 + nbs2;         // v_pk_fma_f16
                b[2 * v]     = res[0];
                b[2 * v + 1] = res[1];
            }

            acc = __builtin_amdgcn_wmma_f32_16x16x32_f16(
                false, a, false, b, (short)0, acc, false, false);
        }
    }

    // Deterministic cross-wave reduction (fixed order, no float atomics).
#pragma unroll
    for (int i = 0; i < 8; ++i) red[wave * 256 + lane * 8 + i] = acc[i];
    __syncthreads();

    // Each thread owns one of the 256 output elements of this 16x16 tile.
    const int e  = tid;                 // element = lane*8 + vgpr
    const int l2 = e >> 3;              // original lane
    const int rr = e & 7;               // original acc VGPR
    float sum = 0.0f;
#pragma unroll
    for (int w = 0; w < 8; ++w) sum += red[w * 256 + e];   // conflict-free reads

    const int row = rr + ((l2 >> 4) << 3);   // D layout: M = vgpr + 8*(lane>=16)
    const int col = n0 + (l2 & 15);          // N = lane % 16
    out[(size_t)row * N_DIM + col] = sum;
}

extern "C" void kernel_launch(void* const* d_in, const int* in_sizes, int n_in,
                              void* d_out, int out_size, void* d_ws, size_t ws_size,
                              hipStream_t stream) {
    const float* x       = (const float*)d_in[0];
    const int*   qweight = (const int*)d_in[1];
    const float* scales  = (const float*)d_in[2];
    const int*   qzeros  = (const int*)d_in[3];

    _Float16* Apack = (_Float16*)d_ws;   // needs 16*4096*2 = 128 KB of scratch

    apack_kernel<<<(NTILES * 32 + 127) / 128, 128, 0, stream>>>(x, Apack);
    qgemm_wmma<<<N_DIM / 16, 256, 0, stream>>>(Apack, qweight, scales, qzeros,
                                               (float*)d_out);
}
